// MultiHeadAttention_41369124995639
// MI455X (gfx1250) — matrix operand order repaired and checked by reading, not compiled, not run on hardware
//
#include <hip/hip_runtime.h>
#include <hip/hip_bf16.h>

typedef __attribute__((ext_vector_type(16))) _Float16 v16h;
typedef __attribute__((ext_vector_type(8)))  _Float16 v8h;
typedef __attribute__((ext_vector_type(8)))  float    v8f;

constexpr int Bc  = 4;
constexpr int Sc  = 2048;
constexpr int Hc  = 1024;
constexpr int NHc = 16;
constexpr int HDc = 64;
constexpr int Mtot = Bc * Sc; // 8192

__device__ __forceinline__ v16h cat16(v8h lo, v8h hi) {
  return __builtin_shufflevector(lo, hi, 0,1,2,3,4,5,6,7,8,9,10,11,12,13,14,15);
}
__device__ __forceinline__ v16h load_frag(const _Float16* p0, const _Float16* p1) {
  return cat16(*(const v8h*)p0, *(const v8h*)p1);
}
__device__ __forceinline__ v8f wmma_f16(v16h a, v16h b, v8f c) {
  // D = A*B + C, f32 accumulate, 16x16x32
  return __builtin_amdgcn_wmma_f32_16x16x32_f16(false, a, false, b, (short)0, c, false, false);
}

// ---------------- conversion kernels ----------------

__global__ void f32_to_f16_kernel(const float* __restrict__ in, _Float16* __restrict__ out, size_t n) {
  size_t i = (size_t)blockIdx.x * blockDim.x + threadIdx.x;
  if (i < n) out[i] = (_Float16)in[i];
}

// w [K=H][N=H] f32 -> wt [N][K] f16 (transposed)
__global__ void w_transpose_f16_kernel(const float* __restrict__ w, _Float16* __restrict__ wt) {
  int c = threadIdx.x & 15;   // k within tile
  int r = threadIdx.x >> 4;   // n within tile
  int k = blockIdx.x * 16 + c;
  int n = blockIdx.y * 16 + r;
  wt[(size_t)n * Hc + k] = (_Float16)w[(size_t)k * Hc + n];
}

// ---------------- WMMA GEMM ----------------
// C[M,N] = A[M,K=H] * W[K,N] + bias[N], with W supplied transposed (Wt[N][K], f16).
// Each wave computes a 32x64 strip: 2 A-frags x 4 B-frags -> 8 WMMAs per 32-K step.
// MODE 0: write f16 Q layout [B,NH,S,HD]
// MODE 1: write f16 K layout [B,NH,S,HD]
// MODE 2: write f16 V-transposed layout [B,NH,HD,S]
// MODE 3: write f32 row-major [M,H] (final output)
template <int MODE>
__global__ __launch_bounds__(256) void gemm_wmma_kernel(
    const _Float16* __restrict__ A, const _Float16* __restrict__ Wt,
    const float* __restrict__ bias, _Float16* __restrict__ o16, float* __restrict__ o32) {
  constexpr int K = Hc;
  const int lane = threadIdx.x & 31;
  const int wave = threadIdx.x >> 5;      // 0..7
  const int l15  = lane & 15;
  const int hi   = lane >> 4;             // 0 or 1
  const int kg   = hi * 8;                // A-frag K group base
  const int m0 = blockIdx.x * 64 + (wave & 1) * 32;   // 32 rows per wave
  const int n0 = blockIdx.y * 256 + (wave >> 1) * 64; // 64 cols per wave

  const _Float16* arow0 = A + (size_t)(m0 + l15) * K;
  const _Float16* arow1 = A + (size_t)(m0 + 16 + l15) * K;
  v8f c[2][4] = {};

  for (int kk = 0; kk < K; kk += 32) {
    // A fragments: lanes 0-15 K kk..kk+7 / kk+16..+23 ; lanes 16-31 +8
    v16h a0 = load_frag(arow0 + kk + kg, arow0 + kk + 16 + kg);
    v16h a1 = load_frag(arow1 + kk + kg, arow1 + kk + 16 + kg);
#pragma unroll
    for (int t = 0; t < 4; ++t) {
      // B fragment: column n = tile + l15, K kk + hi*8 .. +7 / kk+16+hi*8 .. +7 (Wt row)
      const _Float16* br = Wt + (size_t)(n0 + t * 16 + l15) * K + kk + hi * 8;
      v16h b = load_frag(br, br + 16);
      c[0][t] = wmma_f16(a0, b, c[0][t]);
      c[1][t] = wmma_f16(a1, b, c[1][t]);
    }
  }

#pragma unroll
  for (int t = 0; t < 4; ++t) {
    const int col = n0 + t * 16 + l15;
    const float bv = bias[col];
#pragma unroll
    for (int mi = 0; mi < 2; ++mi) {
#pragma unroll
      for (int r = 0; r < 8; ++r) {
        const int row = m0 + mi * 16 + hi * 8 + r;  // C layout: lanes16-31 hold M=8+r
        float v = c[mi][t][r] + bv;
        if constexpr (MODE == 3) {
          o32[(size_t)row * Hc + col] = v;
        } else {
          const int bb = row >> 11;        // / S
          const int s  = row & (Sc - 1);
          const int hh = col >> 6;         // / HD
          const int d  = col & (HDc - 1);
          size_t idx;
          if constexpr (MODE == 2) idx = (((size_t)bb * NHc + hh) * HDc + d) * Sc + s;
          else                     idx = (((size_t)bb * NHc + hh) * Sc + s) * HDc + d;
          o16[idx] = (_Float16)v;
        }
      }
    }
  }
}

// ---------------- flash attention ----------------
// Q,K: [B,NH,S,HD] f16 ; Vt: [B,NH,HD,S] f16 ; ctx out: [B,S,H] f16
// Per wave: 16 queries, 64-key blocks -> 8 score WMMAs + 8 PV WMMAs per softmax.
__global__ __launch_bounds__(128) void flash_attn_kernel(
    const _Float16* __restrict__ Q, const _Float16* __restrict__ Km,
    const _Float16* __restrict__ Vt, _Float16* __restrict__ ctx) {
  __shared__ __align__(16) _Float16 Pl[4][16 * 64];
  const int lane = threadIdx.x & 31;
  const int wave = threadIdx.x >> 5;  // 0..3
  const int l15  = lane & 15;
  const int hi   = lane >> 4;
  const int kg   = hi * 8;
  const int b = blockIdx.z, h = blockIdx.y;
  const int m0 = blockIdx.x * 64 + wave * 16;

  const _Float16* Qb = Q  + (size_t)(b * NHc + h) * Sc * HDc;
  const _Float16* Kb = Km + (size_t)(b * NHc + h) * Sc * HDc;
  const _Float16* Vb = Vt + (size_t)(b * NHc + h) * HDc * Sc;

  // Q A-fragments for K-steps 0 and 32 (HD=64), resident for whole loop
  const _Float16* qrow = Qb + (size_t)(m0 + l15) * HDc;
  v16h qa0 = load_frag(qrow + kg,      qrow + 16 + kg);
  v16h qa1 = load_frag(qrow + 32 + kg, qrow + 48 + kg);

  float mrow[8], lrow[8];
#pragma unroll
  for (int r = 0; r < 8; ++r) { mrow[r] = -1e30f; lrow[r] = 0.0f; }
  v8f o[4] = {};

  _Float16* P = &Pl[wave][0];
  const float scale = 0.125f;  // 1/sqrt(64)

  for (int j0 = 0; j0 < Sc; j0 += 64) {
    if (j0 + 64 < Sc) {  // uniform branch: EXEC stays all-ones for WMMA
      __builtin_prefetch(Kb + (size_t)(j0 + 64 + l15) * HDc, 0, 1);
      __builtin_prefetch(Vb + (size_t)l15 * Sc + j0 + 64, 0, 1);
    }

    // scores: 16 queries x 64 keys = four 16x16 tiles, K-dim = HD = 64
    v8f s[4] = {};
#pragma unroll
    for (int nt = 0; nt < 4; ++nt) {
      const _Float16* kr = Kb + (size_t)(j0 + nt * 16 + l15) * HDc + hi * 8;
      s[nt] = wmma_f16(qa0, load_frag(kr,      kr + 16),  s[nt]);
      s[nt] = wmma_f16(qa1, load_frag(kr + 32, kr + 48), s[nt]);
    }

    // online softmax; rows live in one VGPR across each 16-lane half-wave
    float alpha[8];
#pragma unroll
    for (int r = 0; r < 8; ++r) {
      float a0 = s[0][r] * scale;
      float a1 = s[1][r] * scale;
      float a2 = s[2][r] * scale;
      float a3 = s[3][r] * scale;
      float mx = fmaxf(fmaxf(a0, a1), fmaxf(a2, a3));
#pragma unroll
      for (int d = 1; d < 16; d <<= 1) mx = fmaxf(mx, __shfl_xor(mx, d, 32));
      float mn = fmaxf(mrow[r], mx);
      float al = __expf(mrow[r] - mn);
      float p0 = __expf(a0 - mn);
      float p1 = __expf(a1 - mn);
      float p2 = __expf(a2 - mn);
      float p3 = __expf(a3 - mn);
      float rs = (p0 + p1) + (p2 + p3);
#pragma unroll
      for (int d = 1; d < 16; d <<= 1) rs += __shfl_xor(rs, d, 32);
      mrow[r] = mn;
      lrow[r] = lrow[r] * al + rs;
      alpha[r] = al;
      // stage P into wave-private LDS tile [16][64] (wave-synchronous, DS in-order)
      const int ro = (hi * 8 + r) * 64 + l15;
      P[ro]      = (_Float16)p0;
      P[ro + 16] = (_Float16)p1;
      P[ro + 32] = (_Float16)p2;
      P[ro + 48] = (_Float16)p3;
    }

#pragma unroll
    for (int nt = 0; nt < 4; ++nt)
#pragma unroll
      for (int r = 0; r < 8; ++r) o[nt][r] *= alpha[r];

    // P A-fragments from LDS (K=64 -> two WMMA K-steps)
    const _Float16* pr = P + (size_t)l15 * 64;
    v16h pa0 = load_frag(pr + kg,      pr + 16 + kg);
    v16h pa1 = load_frag(pr + 32 + kg, pr + 48 + kg);

    // O += P(16x64) @ V(64x64): 4 N-tiles x 2 K-steps
#pragma unroll
    for (int nt = 0; nt < 4; ++nt) {
      const _Float16* vr = Vb + (size_t)(nt * 16 + l15) * Sc + j0 + hi * 8;
      o[nt] = wmma_f16(pa0, load_frag(vr,      vr + 16),  o[nt]);
      o[nt] = wmma_f16(pa1, load_frag(vr + 32, vr + 48), o[nt]);
    }
  }

  // epilogue: divide by row sums, write ctx [B,S,H] f16
#pragma unroll
  for (int nt = 0; nt < 4; ++nt) {
    const int col = h * HDc + nt * 16 + l15;
#pragma unroll
    for (int r = 0; r < 8; ++r) {
      const int row = m0 + hi * 8 + r;
      float v = o[nt][r] / lrow[r];
      ctx[((size_t)b * Sc + row) * Hc + col] = (_Float16)v;
    }
  }
}

// ---------------- host launcher ----------------

extern "C" void kernel_launch(void* const* d_in, const int* in_sizes, int n_in,
                              void* d_out, int out_size, void* d_ws, size_t ws_size,
                              hipStream_t stream) {
  (void)in_sizes; (void)n_in; (void)out_size; (void)ws_size;
  const float* x  = (const float*)d_in[0];
  const float* wq = (const float*)d_in[1];
  const float* bq = (const float*)d_in[2];
  const float* wk = (const float*)d_in[3];
  const float* bk = (const float*)d_in[4];
  const float* wv = (const float*)d_in[5];
  const float* bv = (const float*)d_in[6];
  const float* wo = (const float*)d_in[7];
  const float* bo = (const float*)d_in[8];
  float* out = (float*)d_out;

  // workspace layout (f16): x16 16M, 4x wT 2M, q 16M, k 16M, vT 16M, ctx 16M = 88 MiB
  char* p = (char*)d_ws;
  _Float16* x16  = (_Float16*)p; p += (size_t)Mtot * Hc * 2;
  _Float16* wqT  = (_Float16*)p; p += (size_t)Hc * Hc * 2;
  _Float16* wkT  = (_Float16*)p; p += (size_t)Hc * Hc * 2;
  _Float16* wvT  = (_Float16*)p; p += (size_t)Hc * Hc * 2;
  _Float16* woT  = (_Float16*)p; p += (size_t)Hc * Hc * 2;
  _Float16* q16  = (_Float16*)p; p += (size_t)Bc * NHc * Sc * HDc * 2;
  _Float16* k16  = (_Float16*)p; p += (size_t)Bc * NHc * Sc * HDc * 2;
  _Float16* vt16 = (_Float16*)p; p += (size_t)Bc * NHc * HDc * Sc * 2;
  _Float16* ctx16 = (_Float16*)p;

  const size_t nx = (size_t)Mtot * Hc;
  f32_to_f16_kernel<<<(unsigned)((nx + 255) / 256), 256, 0, stream>>>(x, x16, nx);

  dim3 tg(Hc / 16, Hc / 16);
  w_transpose_f16_kernel<<<tg, 256, 0, stream>>>(wq, wqT);
  w_transpose_f16_kernel<<<tg, 256, 0, stream>>>(wk, wkT);
  w_transpose_f16_kernel<<<tg, 256, 0, stream>>>(wv, wvT);
  w_transpose_f16_kernel<<<tg, 256, 0, stream>>>(wo, woT);

  dim3 gg(Mtot / 64, Hc / 256);
  gemm_wmma_kernel<0><<<gg, 256, 0, stream>>>(x16, wqT, bq, q16, nullptr);
  gemm_wmma_kernel<1><<<gg, 256, 0, stream>>>(x16, wkT, bk, k16, nullptr);
  gemm_wmma_kernel<2><<<gg, 256, 0, stream>>>(x16, wvT, bv, vt16, nullptr);

  dim3 fa(Sc / 64, NHc, Bc);
  flash_attn_kernel<<<fa, 128, 0, stream>>>(q16, k16, vt16, ctx16);

  gemm_wmma_kernel<3><<<gg, 256, 0, stream>>>(ctx16, woT, bo, nullptr, out);
}
